// SupProtoConLoss_65644280152913
// MI455X (gfx1250) — matrix operand order repaired and checked by reading, not compiled, not run on hardware
//
#include <hip/hip_runtime.h>
#include <hip/hip_bf16.h>
#include <math.h>

#define NROWS 8192
#define DDIM  512
#define EPS   1e-8f
#define TEMP_INV 10.0f

typedef _Float16 v16h __attribute__((ext_vector_type(16)));
typedef _Float16 v8h  __attribute__((ext_vector_type(8)));
typedef float    v8f  __attribute__((ext_vector_type(8)));

// ---------------------------------------------------------------------------
// Pass 1: row L2-normalize, store as f16 (cos(i,j) = dot of normalized rows;
// norms ~ sqrt(512) >> sqrt(COS_EPS), so per-row clamp is equivalent).
// ---------------------------------------------------------------------------
__global__ void __launch_bounds__(128)
normalize_rows(const float* __restrict__ reps, _Float16* __restrict__ Rh) {
  __shared__ float red[128];
  const int row = blockIdx.x;
  const int tid = threadIdx.x;
  const float* src = reps + (size_t)row * DDIM;
  float ss = 0.f;
  #pragma unroll
  for (int j = tid; j < DDIM; j += 128) { float x = src[j]; ss += x * x; }
  red[tid] = ss;
  __syncthreads();
  for (int off = 64; off > 0; off >>= 1) {
    if (tid < off) red[tid] += red[tid + off];
    __syncthreads();
  }
  const float inv = rsqrtf(fmaxf(red[0], 1e-16f));
  _Float16* dst = Rh + (size_t)row * DDIM;
  #pragma unroll
  for (int j = tid; j < DDIM; j += 128) dst[j] = (_Float16)(src[j] * inv);
}

// ---------------------------------------------------------------------------
// Pass 2: Gram matrix via WMMA + fused contrastive statistics.
// Block = 128 threads = 4 waves. Each wave owns a 32x64 strip:
// 2 A fragments x 4 B fragments -> 8 v_wmma_f32_16x16x32_f16 per K-chunk
// (1.5 b128 loads per WMMA; every B fragment feeds 2 WMMAs).
// Per-row stats: sumSP = sum(s*pos*offdiag), Cc = sum(pos*offdiag),
// Uu = sum(exp(s)*(1-pos)); global max of s via atomicMax on uint bits (s>=0).
// ---------------------------------------------------------------------------
__global__ void __launch_bounds__(128)
gram_stats(const _Float16* __restrict__ Rh,
           const long long* __restrict__ labels,
           float* __restrict__ sumSP, float* __restrict__ Cc,
           float* __restrict__ Uu, unsigned int* __restrict__ Mb) {
  const int lane = threadIdx.x & 31;
  const int wave = threadIdx.x >> 5;
  const int half = lane >> 4;   // K-group select per ISA layout
  const int lrow = lane & 15;   // A: M row / B: N column within tile

  const int mBase = blockIdx.y * 128 + wave * 32;
  const int nBase = blockIdx.x * 64;

  // A 16x32, this code's contiguous k order (A and B alike): lane<16 holds K 0..15, lane>=16 holds
  // K 16..31 of row M=lrow.  B 32x16: lane<16 holds K 0..15 of col
  // N=lrow, lane>=16 holds K 16..31.
  const _Float16* aRow0 = Rh + (size_t)(mBase + lrow) * DDIM;
  const _Float16* aRow1 = Rh + (size_t)(mBase + 16 + lrow) * DDIM;
  const _Float16* bRow  = Rh + (size_t)(nBase + lrow) * DDIM + half * 16;

  v8f acc[2][4];
  #pragma unroll
  for (int u = 0; u < 2; ++u)
    #pragma unroll
    for (int t = 0; t < 4; ++t) acc[u][t] = v8f{};

  for (int k0 = 0; k0 < DDIM; k0 += 32) {
    v16h a0, a1;
    {
      v8h lo = *(const v8h*)(aRow0 + k0 + half * 16);
      v8h hi = *(const v8h*)(aRow0 + k0 + half * 16 + 8);
      #pragma unroll
      for (int i = 0; i < 8; ++i) { a0[i] = lo[i]; a0[8 + i] = hi[i]; }
    }
    {
      v8h lo = *(const v8h*)(aRow1 + k0 + half * 16);
      v8h hi = *(const v8h*)(aRow1 + k0 + half * 16 + 8);
      #pragma unroll
      for (int i = 0; i < 8; ++i) { a1[i] = lo[i]; a1[8 + i] = hi[i]; }
    }

    #pragma unroll
    for (int t = 0; t < 4; ++t) {
      v16h b = *(const v16h*)(bRow + (size_t)(t * 16) * DDIM + k0);
      acc[0][t] = __builtin_amdgcn_wmma_f32_16x16x32_f16(
          false, a0, false, b, (short)0, acc[0][t], false, false);
      acc[1][t] = __builtin_amdgcn_wmma_f32_16x16x32_f16(
          false, a1, false, b, (short)0, acc[1][t], false, false);
    }
  }

  // C layout (32-bit 16x16): VGPR v, lanes 0-15 -> M=v, lanes 16-31 -> M=v+8,
  // N = lrow.  Rows owned by this lane in subtile u: rbase(u)..rbase(u)+7.
  float smax = 0.f;

  #pragma unroll
  for (int u = 0; u < 2; ++u) {
    const int rbase = mBase + u * 16 + half * 8;
    long long rl[8];
    #pragma unroll
    for (int v = 0; v < 8; ++v) rl[v] = labels[rbase + v];

    float sp[8], cn[8], uu[8];
    #pragma unroll
    for (int v = 0; v < 8; ++v) { sp[v] = 0.f; cn[v] = 0.f; uu[v] = 0.f; }

    #pragma unroll
    for (int t = 0; t < 4; ++t) {
      const int col = nBase + t * 16 + lrow;
      const long long cl = labels[col];
      #pragma unroll
      for (int v = 0; v < 8; ++v) {
        const int row = rbase + v;
        float s = ((acc[u][t][v] + 1.0f) * 0.5f + EPS) * TEMP_INV;
        const bool diag = (row == col);
        if (diag) s = 0.f;                 // scores * offdiag
        if (rl[v] == cl) {                 // pos_mask (includes diagonal)
          if (!diag) { sp[v] += s; cn[v] += 1.f; }
        } else {
          uu[v] += __expf(s);              // 1 - pos_mask
        }
        smax = fmaxf(smax, s);
      }
    }

    // Reduce stats over the 16 lanes of each N-group (xor masks 1..8 stay
    // inside a 16-lane half on wave32).
    #pragma unroll
    for (int m = 1; m <= 8; m <<= 1) {
      #pragma unroll
      for (int v = 0; v < 8; ++v) {
        sp[v] += __shfl_xor(sp[v], m, 32);
        cn[v] += __shfl_xor(cn[v], m, 32);
        uu[v] += __shfl_xor(uu[v], m, 32);
      }
    }
    if (lrow == 0) {
      #pragma unroll
      for (int v = 0; v < 8; ++v) {
        atomicAdd(&sumSP[rbase + v], sp[v]);
        atomicAdd(&Cc[rbase + v],    cn[v]);
        atomicAdd(&Uu[rbase + v],    uu[v]);
      }
    }
  }

  #pragma unroll
  for (int m = 1; m <= 16; m <<= 1)
    smax = fmaxf(smax, __shfl_xor(smax, m, 32));
  if (lane == 0) atomicMax(Mb, __float_as_uint(smax));  // s >= 0: uint order ok
}

// ---------------------------------------------------------------------------
// Pass 3: per-row loss, masked mean.  Max-shift applied algebraically:
//   pos   = (sumSP - M*Cc) / (Cc + EPS)
//   neg   = Uu * exp(-M)
//   loss  = -pos + log(neg + EPS); mean over loss > 0
// ---------------------------------------------------------------------------
__global__ void __launch_bounds__(256)
finalize(const float* __restrict__ sumSP, const float* __restrict__ Cc,
         const float* __restrict__ Uu, const unsigned int* __restrict__ Mb,
         float* __restrict__ out) {
  __shared__ float sSum[256];
  __shared__ float sCnt[256];
  const int tid = threadIdx.x;
  const float M = __uint_as_float(*Mb);
  const float eM = __expf(-M);
  float ls = 0.f, lc = 0.f;
  for (int i = tid; i < NROWS; i += 256) {
    const float pos  = (sumSP[i] - M * Cc[i]) / (Cc[i] + EPS);
    const float neg  = Uu[i] * eM;
    const float loss = -pos + __logf(neg + EPS);
    if (loss > 0.f) { ls += loss; lc += 1.f; }
  }
  sSum[tid] = ls; sCnt[tid] = lc;
  __syncthreads();
  for (int off = 128; off > 0; off >>= 1) {
    if (tid < off) { sSum[tid] += sSum[tid + off]; sCnt[tid] += sCnt[tid + off]; }
    __syncthreads();
  }
  if (tid == 0) out[0] = sSum[0] / (sCnt[0] + EPS);
}

// ---------------------------------------------------------------------------
extern "C" void kernel_launch(void* const* d_in, const int* in_sizes, int n_in,
                              void* d_out, int out_size, void* d_ws, size_t ws_size,
                              hipStream_t stream) {
  const float*     reps   = (const float*)d_in[0];
  const long long* labels = (const long long*)d_in[1];
  float* out = (float*)d_out;

  char* ws = (char*)d_ws;
  _Float16* Rh      = (_Float16*)ws;                          // N*D*2 = 8 MB
  float*    sumSP   = (float*)(ws + (size_t)NROWS * DDIM * 2);
  float*    Cc      = sumSP + NROWS;
  float*    Uu      = Cc + NROWS;
  unsigned int* Mb  = (unsigned int*)(Uu + NROWS);

  hipMemsetAsync(sumSP, 0, (3 * NROWS + 1) * sizeof(float), stream);

  normalize_rows<<<NROWS, 128, 0, stream>>>(reps, Rh);

  dim3 grid(NROWS / 64, NROWS / 128);
  gram_stats<<<grid, 128, 0, stream>>>(Rh, labels, sumSP, Cc, Uu, Mb);

  finalize<<<1, 256, 0, stream>>>(sumSP, Cc, Uu, Mb, out);
}
